// SentenceGCN_82257213653075
// MI455X (gfx1250) — matrix operand order repaired and checked by reading, not compiled, not run on hardware
//
#include <hip/hip_runtime.h>
#include <hip/hip_bf16.h>

// ---- types ----
typedef __attribute__((ext_vector_type(16))) __bf16 v16bf;
typedef __attribute__((ext_vector_type(8)))  __bf16 v8bf;
typedef __attribute__((ext_vector_type(4)))  __bf16 v4bf;
typedef __attribute__((ext_vector_type(8)))  float  v8f;

#define NN   50000
#define NE   800000
#define DIM  128

static __device__ __forceinline__ void atomicAddF(float* p, float v) {
  // relaxed, agent scope, result unused -> non-returning global_atomic_add_f32
  __hip_atomic_fetch_add(p, v, __ATOMIC_RELAXED, __HIP_MEMORY_SCOPE_AGENT);
}

// ---- degree / normalization ----
__global__ void k_deg_init(float* __restrict__ deg) {
  int i = blockIdx.x * 256 + threadIdx.x;
  if (i < NN) deg[i] = 1.0f;              // self-loop
}

__global__ void k_deg_count(const int* __restrict__ dst, float* __restrict__ deg) {
  int e = blockIdx.x * 256 + threadIdx.x;
  if (e < NE) atomicAddF(&deg[dst[e]], 1.0f);
}

__global__ void k_rsqrt(float* __restrict__ deg) {
  int i = blockIdx.x * 256 + threadIdx.x;
  if (i < NN) deg[i] = rsqrtf(deg[i]);
}

// ---- f32 -> bf16 conversion (4 elems/thread) ----
__global__ void k_f32_to_bf16(const float* __restrict__ src, __bf16* __restrict__ dst, int n4) {
  int i = blockIdx.x * 256 + threadIdx.x;
  if (i >= n4) return;
  float4 v = ((const float4*)src)[i];
  v4bf o;
  o[0] = (__bf16)v.x; o[1] = (__bf16)v.y; o[2] = (__bf16)v.z; o[3] = (__bf16)v.w;
  ((v4bf*)dst)[i] = o;
}

// ---- W (128x128 row-major f32) -> column-major bf16 ----
__global__ void k_w_transpose_bf16(const float* __restrict__ W, __bf16* __restrict__ Wt) {
  int idx = blockIdx.x * 256 + threadIdx.x;   // 16384 elements
  int k = idx >> 7, n = idx & 127;
  Wt[n * DIM + k] = (__bf16)W[k * DIM + n];
}

// ---- GEMM: C[M x 128] = A(bf16, row-major) @ B (Bt = column-major bf16) ----
// one wave per 16x16 tile; blockDim=256 -> 8 waves cover the 8 column tiles
__global__ void __launch_bounds__(256)
k_gemm_bf16_wmma(const __bf16* __restrict__ A, const __bf16* __restrict__ Bt,
                 float* __restrict__ C) {
  const int lane = threadIdx.x & 31;
  const int wave = threadIdx.x >> 5;
  const int m0   = blockIdx.x * 16;
  const int n0   = wave * 16;
  const int r16  = lane & 15;
  const int half = lane >> 4;                 // K-half select per ISA A/B layouts

  const __bf16* arow = A  + (size_t)(m0 + r16) * DIM;   // A row for this lane
  const __bf16* bcol = Bt + (size_t)(n0 + r16) * DIM;   // B column for this lane

  v8f acc = {};
#pragma unroll
  for (int k = 0; k < DIM; k += 32) {
    // A 16x32 bf16: lanes 0-15 hold K=[k..k+7],[k+16..k+23]; lanes 16-31 the other halves
    v8bf alo = *(const v8bf*)(arow + k + half * 8);
    v8bf ahi = *(const v8bf*)(arow + k + 16 + half * 8);
    v16bf a = __builtin_shufflevector(alo, ahi, 0,1,2,3,4,5,6,7,8,9,10,11,12,13,14,15);
    // B 32x16 bf16 (column-major in memory): lane holds K=[k+half*8..+7],[k+16+half*8..+7] for its column
    v8bf blo = *(const v8bf*)(bcol + k + half * 8);
    v8bf bhi = *(const v8bf*)(bcol + k + 16 + half * 8);
    v16bf b = __builtin_shufflevector(blo, bhi, 0,1,2,3,4,5,6,7,8,9,10,11,12,13,14,15);
    acc = __builtin_amdgcn_wmma_f32_16x16x32_bf16(
        /*neg_a=*/false, a, /*neg_b=*/false, b,
        /*c_mod=*/(short)0, acc, /*reuse_a=*/false, /*reuse_b=*/false);
  }
  // D layout: VGPR r -> M = m0 + half*8 + r, N = n0 + r16
  float* cp = C + (size_t)(m0 + half * 8) * DIM + n0 + r16;
#pragma unroll
  for (int r = 0; r < 8; ++r) cp[(size_t)r * DIM] = acc[r];
}

// ---- out = inv[i]^2 * h[i]  (self-loop term; also initializes accumulator) ----
__global__ void k_self_init(const float* __restrict__ h, const float* __restrict__ inv,
                            float* __restrict__ out) {
  int i = blockIdx.x * 256 + threadIdx.x;     // over N*32 float4s
  if (i >= NN * 32) return;
  float c = inv[i >> 5]; c = c * c;
  float4 v = ((const float4*)h)[i];
  v.x *= c; v.y *= c; v.z *= c; v.w *= c;
  ((float4*)out)[i] = v;
}

// ---- edge scatter: out[dst] += inv[src]*inv[dst] * h[src]; one wave per edge ----
__global__ void __launch_bounds__(256)
k_edge_scatter(const int* __restrict__ src, const int* __restrict__ dst,
               const float* __restrict__ inv, const float* __restrict__ h,
               float* __restrict__ out) {
  int e = blockIdx.x * 8 + (threadIdx.x >> 5);
  if (e >= NE) return;
  int lane = threadIdx.x & 31;
  int s = src[e], d = dst[e];
  float c = inv[s] * inv[d];
  float4 v = ((const float4*)(h + (size_t)s * DIM))[lane];
  float* op = out + (size_t)d * DIM + lane * 4;
  atomicAddF(op + 0, c * v.x);
  atomicAddF(op + 1, c * v.y);
  atomicAddF(op + 2, c * v.z);
  atomicAddF(op + 3, c * v.w);
}

// ---- out = relu(out + b) ----
__global__ void k_bias_relu(float* __restrict__ out, const float* __restrict__ b) {
  int i = blockIdx.x * 256 + threadIdx.x;
  if (i >= NN * DIM) return;
  float v = out[i] + b[i & 127];
  out[i] = v > 0.0f ? v : 0.0f;
}

// ---- mean pool helpers ----
__global__ void k_zero_g(float* __restrict__ g) {
  if (threadIdx.x < DIM) g[threadIdx.x] = 0.0f;
}

__global__ void __launch_bounds__(128)
k_pool(const float* __restrict__ x, float* __restrict__ g) {
  int f = threadIdx.x;
  float s = 0.0f;
  for (int r = blockIdx.x; r < NN; r += gridDim.x)
    s += x[(size_t)r * DIM + f];
  atomicAddF(&g[f], s);
}

// ---- classifier + log_softmax (2 classes) ----
__global__ void __launch_bounds__(128)
k_final(const float* __restrict__ g, const float* __restrict__ Wc,
        const float* __restrict__ bc, float* __restrict__ out) {
  __shared__ float s0[DIM], s1[DIM];
  int f = threadIdx.x;
  float gf = g[f] * (1.0f / (float)NN);
  s0[f] = gf * Wc[f * 2 + 0];
  s1[f] = gf * Wc[f * 2 + 1];
  __syncthreads();
  for (int st = 64; st > 0; st >>= 1) {
    if (f < st) { s0[f] += s0[f + st]; s1[f] += s1[f + st]; }
    __syncthreads();
  }
  if (f == 0) {
    float l0 = s0[0] + bc[0];
    float l1 = s1[0] + bc[1];
    float m = fmaxf(l0, l1);
    float lse = m + logf(expf(l0 - m) + expf(l1 - m));
    out[0] = l0 - lse;
    out[1] = l1 - lse;
  }
}

extern "C" void kernel_launch(void* const* d_in, const int* in_sizes, int n_in,
                              void* d_out, int out_size, void* d_ws, size_t ws_size,
                              hipStream_t stream) {
  (void)in_sizes; (void)n_in; (void)out_size; (void)ws_size;
  const float* x   = (const float*)d_in[0];
  const int*   ei  = (const int*)d_in[1];      // [2, NE] row-major
  const float* W1  = (const float*)d_in[2];
  const float* b1  = (const float*)d_in[3];
  const float* W2  = (const float*)d_in[4];
  const float* b2  = (const float*)d_in[5];
  const float* Wc  = (const float*)d_in[6];
  const float* bc  = (const float*)d_in[7];
  float* outp = (float*)d_out;
  const int* src = ei;
  const int* dst = ei + NE;

  // workspace partition (all chunks 256B-aligned)
  char* w = (char*)d_ws;
  float*  inv = (float*)w;  w += ((NN * 4 + 255) / 256) * 256;        // 200,192 B
  __bf16* Xbf = (__bf16*)w; w += (size_t)NN * DIM * 2;                // 12.8 MB
  __bf16* Wt  = (__bf16*)w; w += DIM * DIM * 2;                       // 32 KB
  float*  h   = (float*)w;  w += (size_t)NN * DIM * 4;                // 25.6 MB
  float*  agg = (float*)w;  w += (size_t)NN * DIM * 4;                // 25.6 MB
  float*  g   = (float*)w;                                           // 512 B

  const int nodeBlk = (NN + 255) / 256;          // 196
  const int edgeBlk = (NE + 255) / 256;          // 3125
  const int vec4Blk = (NN * 32 + 255) / 256;     // 6250
  const int featBlk = (NN * DIM + 255) / 256;    // 25000
  const int gemmBlk = NN / 16;                   // 3125
  const int scatBlk = (NE + 7) / 8;              // 100000

  // normalization coefficients
  k_deg_init <<<nodeBlk, 256, 0, stream>>>(inv);
  k_deg_count<<<edgeBlk, 256, 0, stream>>>(dst, inv);
  k_rsqrt    <<<nodeBlk, 256, 0, stream>>>(inv);

  // ---- layer 1 ----
  k_f32_to_bf16     <<<vec4Blk, 256, 0, stream>>>(x, Xbf, NN * 32);
  k_w_transpose_bf16<<<64, 256, 0, stream>>>(W1, Wt);
  k_gemm_bf16_wmma  <<<gemmBlk, 256, 0, stream>>>(Xbf, Wt, h);
  k_self_init       <<<vec4Blk, 256, 0, stream>>>(h, inv, agg);
  k_edge_scatter    <<<scatBlk, 256, 0, stream>>>(src, dst, inv, h, agg);
  k_bias_relu       <<<featBlk, 256, 0, stream>>>(agg, b1);

  // ---- layer 2 ----
  k_f32_to_bf16     <<<vec4Blk, 256, 0, stream>>>(agg, Xbf, NN * 32);
  k_w_transpose_bf16<<<64, 256, 0, stream>>>(W2, Wt);
  k_gemm_bf16_wmma  <<<gemmBlk, 256, 0, stream>>>(Xbf, Wt, h);
  k_self_init       <<<vec4Blk, 256, 0, stream>>>(h, inv, agg);
  k_edge_scatter    <<<scatBlk, 256, 0, stream>>>(src, dst, inv, h, agg);
  k_bias_relu       <<<featBlk, 256, 0, stream>>>(agg, b2);

  // ---- mean pool + classifier + log_softmax ----
  k_zero_g<<<1, 128, 0, stream>>>(g);
  k_pool  <<<512, 128, 0, stream>>>(agg, g);
  k_final <<<1, 128, 0, stream>>>(g, Wc, bc, outp);
}
